// VectorQuantizer_2851858284842
// MI455X (gfx1250) — matrix operand order repaired and checked by reading, not compiled, not run on hardware
//
#include <hip/hip_runtime.h>
#include <hip/hip_bf16.h>

typedef __attribute__((ext_vector_type(16))) __bf16 v16bf;
typedef __attribute__((ext_vector_type(8)))  float  v8f;

#define N_ROWS 65536
#define K_CB   1024
#define D_DIM  128
#define BETA_F 0.25f

struct __align__(16) Vec128 { unsigned int w[4]; };

union BF16Vec {
    v16bf v;
    unsigned short u[16];
    Vec128 q[2];
};

__device__ __forceinline__ unsigned short f32_to_bf16_rne(float f) {
    unsigned u = __float_as_uint(f);
    return (unsigned short)((u + 0x7FFFu + ((u >> 16) & 1u)) >> 16);
}
__device__ __forceinline__ float bf16u_to_f32(unsigned short h) {
    return __uint_as_float(((unsigned)h) << 16);
}

// ---------------------------------------------------------------------------
// Prep: codebook f32 -> bf16 hi/lo split, per-code squared norm, zero accum.
// grid = K_CB blocks of D_DIM threads (one block per codebook row).
// ---------------------------------------------------------------------------
__global__ void vq_prep(const float* __restrict__ cb,
                        unsigned short* __restrict__ cb_hi,
                        unsigned short* __restrict__ cb_lo,
                        float* __restrict__ norms,
                        float* __restrict__ accum) {
    const int k = blockIdx.x;
    const int d = threadIdx.x;
    const float x = cb[k * D_DIM + d];
    const unsigned short hi = f32_to_bf16_rne(x);
    const unsigned short lo = f32_to_bf16_rne(x - bf16u_to_f32(hi));
    cb_hi[k * D_DIM + d] = hi;
    cb_lo[k * D_DIM + d] = lo;

    float s = x * x;
    #pragma unroll
    for (int off = 16; off > 0; off >>= 1) s += __shfl_down(s, off, 32);
    __shared__ float part[4];
    if ((threadIdx.x & 31) == 0) part[threadIdx.x >> 5] = s;
    __syncthreads();
    if (threadIdx.x == 0) {
        norms[k] = part[0] + part[1] + part[2] + part[3];
        if (k == 0) *accum = 0.0f;
    }
}

// ---------------------------------------------------------------------------
// Main: each block = 8 waves; each wave owns TWO 16-row A tiles (M=32) so a
// single set of B fragments (16 loads) feeds 24 WMMAs -> compute bound.
// Compensated-bf16 WMMA GEMM vs all 1024 codes + running argmin + gather.
// ---------------------------------------------------------------------------
__global__ void __launch_bounds__(256)
vq_main(const float* __restrict__ enc,
        const float* __restrict__ cb,
        const unsigned short* __restrict__ cb_hi,
        const unsigned short* __restrict__ cb_lo,
        const float* __restrict__ norms,
        float* __restrict__ out_idx,
        float* __restrict__ out_q,
        float* __restrict__ accum) {
    const int lane    = threadIdx.x & 31;
    const int wave    = threadIdx.x >> 5;
    const int rowBase = blockIdx.x * 256 + wave * 32;
    const int lrow    = lane & 15;   // row (A) / column (B) within tile
    const int half    = lane >> 4;

    // ---- Load 2 A tiles (each 16 x 128 f32), split into bf16 hi/lo ----
    // 16-bit A layout: lane m = lrow; K elements {b0..b0+7} U {b0+16..b0+23},
    // b0 = half*8, occupying vector slots 0..7 and 8..15.
    BF16Vec a_hi[2][4], a_lo[2][4];
    #pragma unroll
    for (int t = 0; t < 2; ++t) {
        const float* rp = enc + (size_t)(rowBase + t * 16 + lrow) * D_DIM;
        #pragma unroll
        for (int ks = 0; ks < 4; ++ks) {
            const int c0 = ks * 32 + half * 8;
            #pragma unroll
            for (int chunk = 0; chunk < 2; ++chunk) {
                const float4 x0 = *(const float4*)(rp + c0 + chunk * 16);
                const float4 x1 = *(const float4*)(rp + c0 + chunk * 16 + 4);
                const float xs[8] = {x0.x, x0.y, x0.z, x0.w, x1.x, x1.y, x1.z, x1.w};
                #pragma unroll
                for (int i = 0; i < 8; ++i) {
                    const unsigned short h = f32_to_bf16_rne(xs[i]);
                    a_hi[t][ks].u[chunk * 8 + i] = h;
                    a_lo[t][ks].u[chunk * 8 + i] =
                        f32_to_bf16_rne(xs[i] - bf16u_to_f32(h));
                }
            }
        }
    }

    float best[2][8];
    int   bidx[2][8];
    #pragma unroll
    for (int t = 0; t < 2; ++t)
        #pragma unroll
        for (int r = 0; r < 8; ++r) { best[t][r] = 3.4e38f; bidx[t][r] = 0; }

    // ---- Sweep all K_CB codes in 16-wide tiles ----
    for (int kt = 0; kt < K_CB / 16; ++kt) {
        const int ncol = kt * 16 + lrow;
        const float nrm = norms[ncol];
        // 16-bit B layout: lane n = lrow; K = {half*8 + 0..7} U {16 + half*8 + 0..7}.
        const unsigned short* bp_h = cb_hi + (size_t)ncol * D_DIM + half * 8;
        const unsigned short* bp_l = cb_lo + (size_t)ncol * D_DIM + half * 8;

        // prefetch next tile's B fragments into cache (global_prefetch_b8)
        if (kt < K_CB / 16 - 1) {
            __builtin_prefetch(bp_h + 16 * D_DIM, 0, 1);
            __builtin_prefetch(bp_l + 16 * D_DIM, 0, 1);
        }

        v8f acc[2][2] = {};
        #pragma unroll
        for (int ks = 0; ks < 4; ++ks) {
            BF16Vec bh, bl;
            bh.q[0] = *(const Vec128*)(bp_h + ks * 32);
            bh.q[1] = *(const Vec128*)(bp_h + ks * 32 + 16);
            bl.q[0] = *(const Vec128*)(bp_l + ks * 32);
            bl.q[1] = *(const Vec128*)(bp_l + ks * 32 + 16);
            #pragma unroll
            for (int t = 0; t < 2; ++t) {
                v8f& a = acc[t][ks & 1];
                a = __builtin_amdgcn_wmma_f32_16x16x32_bf16(
                    false, a_hi[t][ks].v, false, bh.v, (short)0, a, false, false);
                a = __builtin_amdgcn_wmma_f32_16x16x32_bf16(
                    false, a_lo[t][ks].v, false, bh.v, (short)0, a, false, false);
                a = __builtin_amdgcn_wmma_f32_16x16x32_bf16(
                    false, a_hi[t][ks].v, false, bl.v, (short)0, a, false, false);
            }
        }
        // score = ||c||^2 - 2*dot ; running argmin per residue class
        #pragma unroll
        for (int t = 0; t < 2; ++t)
            #pragma unroll
            for (int r = 0; r < 8; ++r) {
                const float s = fmaf(acc[t][0][r] + acc[t][1][r], -2.0f, nrm);
                const bool take = (s < best[t][r]);
                best[t][r] = take ? s    : best[t][r];
                bidx[t][r] = take ? ncol : bidx[t][r];
            }
    }

    // ---- 16-lane argmin reduction (low-index tie break) ----
    #pragma unroll
    for (int off = 1; off < 16; off <<= 1) {
        #pragma unroll
        for (int t = 0; t < 2; ++t)
            #pragma unroll
            for (int r = 0; r < 8; ++r) {
                const float ov = __shfl_xor(best[t][r], off, 32);
                const int   oi = __shfl_xor(bidx[t][r], off, 32);
                const bool take =
                    (ov < best[t][r]) || (ov == best[t][r] && oi < bidx[t][r]);
                best[t][r] = take ? ov : best[t][r];
                bidx[t][r] = take ? oi : bidx[t][r];
            }
    }
    // slot r of half h now holds the argmin for row (rowBase + t*16 + 8*h + r)
    if (lrow == 0) {
        #pragma unroll
        for (int t = 0; t < 2; ++t)
            #pragma unroll
            for (int r = 0; r < 8; ++r)
                out_idx[rowBase + t * 16 + half * 8 + r] = (float)bidx[t][r];
    }

    // ---- gather codebook rows (f32), write quantized_st, accumulate MSE ----
    float sq = 0.0f;
    const int dbase = lane * 4;
    #pragma unroll
    for (int t = 0; t < 2; ++t) {
        #pragma unroll
        for (int m = 0; m < 16; ++m) {
            const int im  = __shfl(bidx[t][m & 7], (m >> 3) * 16, 32);
            const int row = rowBase + t * 16 + m;
            const float4 c = *(const float4*)(cb  + (size_t)im  * D_DIM + dbase);
            const float4 e = *(const float4*)(enc + (size_t)row * D_DIM + dbase);
            *(float4*)(out_q + (size_t)row * D_DIM + dbase) = c;
            const float dx = c.x - e.x, dy = c.y - e.y;
            const float dz = c.z - e.z, dw = c.w - e.w;
            sq = fmaf(dx, dx, sq); sq = fmaf(dy, dy, sq);
            sq = fmaf(dz, dz, sq); sq = fmaf(dw, dw, sq);
        }
    }
    #pragma unroll
    for (int off = 16; off > 0; off >>= 1) sq += __shfl_down(sq, off, 32);
    if (lane == 0) atomicAdd(accum, sq);
}

// ---------------------------------------------------------------------------
// Finalize losses. commitment == embedding == mse numerically (stop_gradient
// does not change values); vq = beta*mse + mse.
// ---------------------------------------------------------------------------
__global__ void vq_final(const float* __restrict__ accum,
                         float* __restrict__ out_losses) {
    const float mse = *accum * (1.0f / ((float)N_ROWS * (float)D_DIM));
    out_losses[0] = mse * BETA_F + mse;  // vq_loss
    out_losses[1] = mse;                 // embedding_loss
    out_losses[2] = mse;                 // commitment_loss
}

extern "C" void kernel_launch(void* const* d_in, const int* in_sizes, int n_in,
                              void* d_out, int out_size, void* d_ws, size_t ws_size,
                              hipStream_t stream) {
    const float* enc = (const float*)d_in[0];   // encoding  (N x D) f32
    const float* cb  = (const float*)d_in[1];   // codebook  (K x D) f32
    float* out = (float*)d_out;

    char* ws = (char*)d_ws;
    unsigned short* cb_hi = (unsigned short*)ws;                  // 256 KB
    unsigned short* cb_lo = (unsigned short*)(ws + (256u << 10)); // 256 KB
    float* norms = (float*)(ws + (512u << 10));                   //   4 KB
    float* accum = (float*)(ws + (512u << 10) + 4096);            //   4 B

    float* out_idx    = out;                                      // N floats
    float* out_q      = out + N_ROWS;                             // N*D floats
    float* out_losses = out + N_ROWS + (size_t)N_ROWS * D_DIM;    // 3 floats

    vq_prep<<<K_CB, D_DIM, 0, stream>>>(cb, cb_hi, cb_lo, norms, accum);
    vq_main<<<N_ROWS / 256, 256, 0, stream>>>(enc, cb, cb_hi, cb_lo, norms,
                                              out_idx, out_q, accum);
    vq_final<<<1, 1, 0, stream>>>(accum, out_losses);
}
